// ShuntedTransformer_83966610637382
// MI455X (gfx1250) — matrix operand order repaired and checked by reading, not compiled, not run on hardware
//
#include <hip/hip_runtime.h>
#include <hip/hip_bf16.h>
#include <cmath>

typedef __attribute__((ext_vector_type(16))) _Float16 v16h;
typedef __attribute__((ext_vector_type(8)))  float    v8f;

#define BB    8
#define CC    128
#define NTOK  4100
#define N0T   4096
#define MSGT  4
#define HWIMG 64
#define QT_TILES 257   // ceil(4100/16)

__device__ __forceinline__ int imin(int a, int b) { return a < b ? a : b; }

// ---------------------------------------------------------------- concat x||msg
__global__ void k_concat_xq(const float* __restrict__ x, const float* __restrict__ msg,
                            float* __restrict__ xq) {
  const size_t total = (size_t)BB * NTOK * CC;
  for (size_t i = blockIdx.x * (size_t)blockDim.x + threadIdx.x; i < total;
       i += (size_t)gridDim.x * blockDim.x) {
    int c = (int)(i % CC);
    size_t r = i / CC;
    int t = (int)(r % NTOK);
    int b = (int)(r / NTOK);
    float v = (t < N0T) ? x[((size_t)b * N0T + t) * CC + c]
                        : msg[((size_t)b * MSGT + (t - N0T)) * CC + c];
    xq[i] = v;
  }
}

// ---------------------------------------------------------------- generic WMMA GEMM
// out[M,N] = A[M,K] @ W[N,K]^T + bias[N].
// One wave computes a 16x64 output tile: one A fragment feeds 4 WMMAs (4x A reuse).
// Requires K % 32 == 0, N % 64 == 0.
__global__ void k_gemm_wmma(const float* __restrict__ A, const float* __restrict__ W,
                            const float* __restrict__ bias, float* __restrict__ out,
                            int M, int N, int K) {
  const int tn = blockIdx.x;      // 64-wide column tile
  const int tm = blockIdx.y;      // 16-high row tile
  const int lane = threadIdx.x;
  const bool hi = lane >= 16;
  const int l16 = lane & 15;

  v8f acc0 = {}, acc1 = {}, acc2 = {}, acc3 = {};
  const float* arow  = A + (size_t)(tm * 16 + l16) * K;
  const float* wrow0 = W + (size_t)(tn * 64 +  0 + l16) * K;
  const float* wrow1 = W + (size_t)(tn * 64 + 16 + l16) * K;
  const float* wrow2 = W + (size_t)(tn * 64 + 32 + l16) * K;
  const float* wrow3 = W + (size_t)(tn * 64 + 48 + l16) * K;
  const int aoff = hi ? 16 : 0;   // A frag: lanes<16 K 0-15; lanes>=16 K 16-31
  const int boff = hi ? 16 : 0;   // B frag: lanes<16 K 0-15; lanes>=16 K 16-31

  for (int k0 = 0; k0 < K; k0 += 32) {
    if (k0 + 32 < K) {
      __builtin_prefetch(arow  + k0 + 32, 0, 3);   // global_prefetch_b8: next A block
      __builtin_prefetch(wrow0 + k0 + 32, 0, 3);
    }
    v16h af;
#pragma unroll
    for (int j = 0; j < 8; ++j) {
      af[j]     = (_Float16)arow[k0 + aoff + j];
      af[j + 8] = (_Float16)arow[k0 + aoff + 8 + j];
    }
    v16h bf0, bf1, bf2, bf3;
#pragma unroll
    for (int j = 0; j < 16; ++j) {
      bf0[j] = (_Float16)wrow0[k0 + boff + j];
      bf1[j] = (_Float16)wrow1[k0 + boff + j];
      bf2[j] = (_Float16)wrow2[k0 + boff + j];
      bf3[j] = (_Float16)wrow3[k0 + boff + j];
    }
    acc0 = __builtin_amdgcn_wmma_f32_16x16x32_f16(false, af, false, bf0, (short)0, acc0, false, false);
    acc1 = __builtin_amdgcn_wmma_f32_16x16x32_f16(false, af, false, bf1, (short)0, acc1, false, false);
    acc2 = __builtin_amdgcn_wmma_f32_16x16x32_f16(false, af, false, bf2, (short)0, acc2, false, false);
    acc3 = __builtin_amdgcn_wmma_f32_16x16x32_f16(false, af, false, bf3, (short)0, acc3, false, false);
  }

  const int mbase = tm * 16 + (hi ? 8 : 0);
  v8f accs[4] = {acc0, acc1, acc2, acc3};
#pragma unroll
  for (int s = 0; s < 4; ++s) {
    const int ncol = tn * 64 + s * 16 + l16;
    const float bv = bias ? bias[ncol] : 0.f;
#pragma unroll
    for (int i = 0; i < 8; ++i)
      out[(size_t)(mbase + i) * N + ncol] = accs[s][i] + bv;
  }
}

// ---------------------------------------------------------------- im2col for SR conv
// Rows per batch: L spatial tokens then MSGT msg rows (msg value replicated over all
// kernel positions -> GEMM reproduces msg @ sr_w.sum((-1,-2)).T exactly).
__global__ void k_im2col(const float* __restrict__ x, const float* __restrict__ msg,
                         float* __restrict__ A, int L, int Lr, int stride, int KK, int ksz) {
  const int Lm = L + MSGT;
  const size_t total = (size_t)BB * Lm * KK;
  const int kk2 = ksz * ksz;
  for (size_t i = blockIdx.x * (size_t)blockDim.x + threadIdx.x; i < total;
       i += (size_t)gridDim.x * blockDim.x) {
    int k = (int)(i % KK);
    size_t r = i / KK;
    int t = (int)(r % Lm);
    int b = (int)(r / Lm);
    int c = k / kk2;
    int j = k % kk2;
    float v;
    if (t < L) {
      int hr = t / Lr, wr = t % Lr;
      int y  = hr * stride + (j / ksz);
      int xx = wr * stride + (j % ksz);
      v = x[((size_t)b * N0T + y * HWIMG + xx) * CC + c];
    } else {
      v = msg[((size_t)b * MSGT + (t - L)) * CC + c];
    }
    A[i] = v;
  }
}

// ---------------------------------------------------------------- LayerNorm + exact GELU
__global__ void k_ln_gelu(const float* __restrict__ in, const float* __restrict__ g,
                          const float* __restrict__ bt, float* __restrict__ out, int rows) {
  __shared__ float sh[CC];
  const int t = threadIdx.x;
  const int r = blockIdx.x;
  if (r >= rows) return;
  float v = in[(size_t)r * CC + t];
  sh[t] = v; __syncthreads();
  for (int s = CC / 2; s > 0; s >>= 1) { if (t < s) sh[t] += sh[t + s]; __syncthreads(); }
  float mu = sh[0] * (1.f / CC); __syncthreads();
  float d = v - mu;
  sh[t] = d * d; __syncthreads();
  for (int s = CC / 2; s > 0; s >>= 1) { if (t < s) sh[t] += sh[t + s]; __syncthreads(); }
  float var = sh[0] * (1.f / CC);
  float y = d * rsqrtf(var + 1e-5f) * g[t] + bt[t];
  out[(size_t)r * CC + t] = 0.5f * y * (1.f + erff(y * 0.70710678118654752f));
}

// ---------------------------------------------------------------- v_new = v + (dwconv(v) | v_msg)
__global__ void k_vconv(const float* __restrict__ kvb, const float* __restrict__ lcw,
                        const float* __restrict__ lcb, float* __restrict__ vnew,
                        int L, int Lr) {
  const int Lm = L + MSGT;
  const size_t total = (size_t)BB * Lm * 64;
  for (size_t i = blockIdx.x * (size_t)blockDim.x + threadIdx.x; i < total;
       i += (size_t)gridDim.x * blockDim.x) {
    int ch = (int)(i % 64);
    size_t r = i / 64;
    int t = (int)(r % Lm);
    int b = (int)(r / Lm);
    float v = kvb[r * CC + 64 + ch];
    float add;
    if (t < L) {
      int hr = t / Lr, wr = t % Lr;
      float s = lcb[ch];
#pragma unroll
      for (int dy = -1; dy <= 1; ++dy)
#pragma unroll
        for (int dx = -1; dx <= 1; ++dx) {
          int y = hr + dy, xx = wr + dx;
          if (y >= 0 && y < Lr && xx >= 0 && xx < Lr)
            s += kvb[((size_t)b * Lm + y * Lr + xx) * CC + 64 + ch] *
                 lcw[ch * 9 + (dy + 1) * 3 + (dx + 1)];
        }
      add = s;
    } else {
      add = v;   // msg tokens bypass the conv (v_add = v)
    }
    vnew[r * 64 + ch] = v + add;
  }
}

// ---------------------------------------------------------------- flash attention
// One wave: (batch b, branch-head hh, 16-query tile). Streams 32-key blocks:
// 2 score WMMAs + online softmax + LDS transpose of P + 2 PV WMMAs.
__global__ void k_attn(const float* __restrict__ qlin, const float* __restrict__ kvb,
                       const float* __restrict__ vnew, float* __restrict__ xcat,
                       int Lm, int headBase, int colBase) {
  int idx = blockIdx.x;
  const int qt = idx % QT_TILES; idx /= QT_TILES;
  const int hh = idx % 2;
  const int b  = idx / 2;
  const int lane = threadIdx.x;
  const bool hi = lane >= 16;
  const int l16 = lane & 15;
  const int q0 = qt * 16;
  const float scale = 0.17677669529663687f;   // 32^-0.5

  __shared__ _Float16 pl[16 * 32];

  // Q fragment (A layout), loaded once
  v16h qa;
  {
    int qr = imin(q0 + l16, NTOK - 1);
    const float* qrow = qlin + ((size_t)b * NTOK + qr) * CC + (headBase + hh) * 32;
    int aoff = hi ? 16 : 0;
#pragma unroll
    for (int j = 0; j < 8; ++j) {
      qa[j]     = (_Float16)qrow[aoff + j];
      qa[j + 8] = (_Float16)qrow[aoff + 8 + j];
    }
  }

  float rm[8], rl[8];
#pragma unroll
  for (int i = 0; i < 8; ++i) { rm[i] = -3.0e38f; rl[i] = 0.f; }
  v8f o0 = {}, o1 = {};

  for (int kb0 = 0; kb0 < Lm; kb0 += 32) {
    // K fragments (B layout): dh=32 along K, 16 keys along N, two key tiles
    const int kd = hi ? 16 : 0;
    const int key1 = kb0 + l16;
    const int key2 = kb0 + 16 + l16;
    const bool val1 = key1 < Lm, val2 = key2 < Lm;
    const float* k1 = kvb + ((size_t)b * Lm + imin(key1, Lm - 1)) * CC + hh * 32 + kd;
    const float* k2 = kvb + ((size_t)b * Lm + imin(key2, Lm - 1)) * CC + hh * 32 + kd;
    if (kb0 + 32 < Lm) {
      __builtin_prefetch(k1 + 32 * CC, 0, 3);   // next key block (same lane pattern)
      __builtin_prefetch(k2 + 32 * CC, 0, 3);
    }
    v16h kf1, kf2;
#pragma unroll
    for (int j = 0; j < 16; ++j) { kf1[j] = (_Float16)k1[j]; kf2[j] = (_Float16)k2[j]; }

    v8f zc = {};
    v8f s1 = __builtin_amdgcn_wmma_f32_16x16x32_f16(false, qa, false, kf1, (short)0, zc, false, false);
    v8f s2 = __builtin_amdgcn_wmma_f32_16x16x32_f16(false, qa, false, kf2, (short)0, zc, false, false);

    float p1[8], p2[8], corr[8];
#pragma unroll
    for (int i = 0; i < 8; ++i) {
      p1[i] = val1 ? s1[i] * scale : -1e30f;
      p2[i] = val2 ? s2[i] * scale : -1e30f;
    }
    // online softmax: row reductions across each 16-lane half (xor 1,2,4,8)
#pragma unroll
    for (int i = 0; i < 8; ++i) {
      float mx = fmaxf(p1[i], p2[i]);
      for (int msk = 1; msk < 16; msk <<= 1) mx = fmaxf(mx, __shfl_xor(mx, msk, 32));
      float mnew = fmaxf(rm[i], mx);
      corr[i] = __expf(rm[i] - mnew);
      rm[i] = mnew;
      p1[i] = __expf(p1[i] - mnew);
      p2[i] = __expf(p2[i] - mnew);
      float rs = p1[i] + p2[i];
      for (int msk = 1; msk < 16; msk <<= 1) rs += __shfl_xor(rs, msk, 32);
      rl[i] = rl[i] * corr[i] + rs;
      o0[i] *= corr[i];
      o1[i] *= corr[i];
    }

    // transpose P (C layout -> A layout) through LDS, as f16
    __syncthreads();
    {
      int prow0 = hi ? 8 : 0;
#pragma unroll
      for (int i = 0; i < 8; ++i) {
        pl[(prow0 + i) * 32 + l16]      = (_Float16)p1[i];
        pl[(prow0 + i) * 32 + 16 + l16] = (_Float16)p2[i];
      }
    }
    __syncthreads();
    v16h pa;
    {
      int aoff = hi ? 16 : 0;
#pragma unroll
      for (int j = 0; j < 8; ++j) {
        pa[j]     = pl[l16 * 32 + aoff + j];
        pa[j + 8] = pl[l16 * 32 + aoff + 8 + j];
      }
    }

    // V fragments (B layout): K=32 keys, N=16 dims; invalid keys clamp (P==0 there)
    const int keyoff = hi ? 16 : 0;
    v16h vb0, vb1;
#pragma unroll
    for (int j = 0; j < 16; ++j) {
      int key = imin(kb0 + keyoff + j, Lm - 1);
      const float* vr = vnew + ((size_t)b * Lm + key) * 64 + hh * 32;
      vb0[j] = (_Float16)vr[l16];
      vb1[j] = (_Float16)vr[16 + l16];
    }
    o0 = __builtin_amdgcn_wmma_f32_16x16x32_f16(false, pa, false, vb0, (short)0, o0, false, false);
    o1 = __builtin_amdgcn_wmma_f32_16x16x32_f16(false, pa, false, vb1, (short)0, o1, false, false);
  }

  // epilogue: normalize and scatter into xcat[:, colBase + hh*32 + d]
  const int prow0 = hi ? 8 : 0;
#pragma unroll
  for (int i = 0; i < 8; ++i) {
    float inv = 1.f / rl[i];
    int qr = q0 + prow0 + i;
    if (qr < NTOK) {
      float* orow = xcat + ((size_t)b * NTOK + qr) * CC + colBase + hh * 32;
      orow[l16]      = o0[i] * inv;
      orow[16 + l16] = o1[i] * inv;
    }
  }
}

// ---------------------------------------------------------------- residual + tuple scatter
__global__ void k_resid_out(const float* __restrict__ projout, const float* __restrict__ xq,
                            float* __restrict__ dout) {
  const size_t total = (size_t)BB * NTOK * CC;
  for (size_t i = blockIdx.x * (size_t)blockDim.x + threadIdx.x; i < total;
       i += (size_t)gridDim.x * blockDim.x) {
    int c = (int)(i % CC);
    size_t r = i / CC;
    int t = (int)(r % NTOK);
    int b = (int)(r / NTOK);
    float v = projout[i] + xq[i];
    size_t dst = (t < N0T)
        ? ((size_t)b * N0T + t) * CC + c
        : (size_t)BB * N0T * CC + ((size_t)b * MSGT + (t - N0T)) * CC + c;
    dout[dst] = v;
  }
}

// ================================================================ host
extern "C" void kernel_launch(void* const* d_in, const int* in_sizes, int n_in,
                              void* d_out, int out_size, void* d_ws, size_t ws_size,
                              hipStream_t stream) {
  (void)in_sizes; (void)n_in; (void)out_size; (void)ws_size;
  const float* x      = (const float*)d_in[0];
  const float* msg    = (const float*)d_in[1];
  const float* Wq     = (const float*)d_in[2];
  const float* bq     = (const float*)d_in[3];
  const float* sr1_w  = (const float*)d_in[4];
  const float* sr1_b  = (const float*)d_in[5];
  const float* ln1_g  = (const float*)d_in[6];
  const float* ln1_b  = (const float*)d_in[7];
  const float* sr2_w  = (const float*)d_in[8];
  const float* sr2_b  = (const float*)d_in[9];
  const float* ln2_g  = (const float*)d_in[10];
  const float* ln2_b  = (const float*)d_in[11];
  const float* kv1_w  = (const float*)d_in[12];
  const float* kv1_b  = (const float*)d_in[13];
  const float* kv2_w  = (const float*)d_in[14];
  const float* kv2_b  = (const float*)d_in[15];
  const float* lc1_w  = (const float*)d_in[16];
  const float* lc1_b  = (const float*)d_in[17];
  const float* lc2_w  = (const float*)d_in[18];
  const float* lc2_b  = (const float*)d_in[19];
  const float* proj_w = (const float*)d_in[20];
  const float* proj_b = (const float*)d_in[21];
  float* out = (float*)d_out;

  float* ws = (float*)d_ws;
  const size_t SZ_XQ = (size_t)BB * NTOK * CC;             // 4,198,400 floats
  float* xq   = ws;
  float* qlin = xq   + SZ_XQ;
  float* xcat = qlin + SZ_XQ;
  float* im2c = xcat + SZ_XQ;                              // branch scratch / proj out
  float* xr   = im2c + (size_t)BB * 260 * 2048;            // im2col max (branch1)
  float* xk   = xr   + (size_t)BB * 1028 * CC;
  float* kvb  = xk   + (size_t)BB * 1028 * CC;
  float* vnew = kvb  + (size_t)BB * 1028 * CC;

  // x_q = concat(x, msg)
  k_concat_xq<<<2048, 256, 0, stream>>>(x, msg, xq);

  // q = x_q @ Wq^T + bq   (M=32800, N=128, K=128)
  {
    dim3 g(CC / 64, (BB * NTOK) / 16);
    k_gemm_wmma<<<g, 32, 0, stream>>>(xq, Wq, bq, qlin, BB * NTOK, CC, CC);
  }

  // ---------------- branch 1: stride 4, L=256, Lm=260, K=2048
  {
    const int L = 256, Lr = 16, stride = 4, ksz = 4, KK = 2048, Lm = 260;
    k_im2col<<<4096, 256, 0, stream>>>(x, msg, im2c, L, Lr, stride, KK, ksz);
    dim3 g(CC / 64, (BB * Lm) / 16);
    k_gemm_wmma<<<g, 32, 0, stream>>>(im2c, sr1_w, sr1_b, xr, BB * Lm, CC, KK);
    k_ln_gelu<<<BB * Lm, CC, 0, stream>>>(xr, ln1_g, ln1_b, xk, BB * Lm);
    k_gemm_wmma<<<g, 32, 0, stream>>>(xk, kv1_w, kv1_b, kvb, BB * Lm, CC, CC);
    k_vconv<<<520, 256, 0, stream>>>(kvb, lc1_w, lc1_b, vnew, L, Lr);
    k_attn<<<BB * 2 * QT_TILES, 32, 0, stream>>>(qlin, kvb, vnew, xcat, Lm, 0, 0);
  }

  // ---------------- branch 2: stride 2, L=1024, Lm=1028, K=512
  {
    const int L = 1024, Lr = 32, stride = 2, ksz = 2, KK = 512, Lm = 1028;
    k_im2col<<<4096, 256, 0, stream>>>(x, msg, im2c, L, Lr, stride, KK, ksz);
    dim3 g(CC / 64, (BB * Lm) / 16);
    k_gemm_wmma<<<g, 32, 0, stream>>>(im2c, sr2_w, sr2_b, xr, BB * Lm, CC, KK);
    k_ln_gelu<<<BB * Lm, CC, 0, stream>>>(xr, ln2_g, ln2_b, xk, BB * Lm);
    k_gemm_wmma<<<g, 32, 0, stream>>>(xk, kv2_w, kv2_b, kvb, BB * Lm, CC, CC);
    k_vconv<<<2056, 256, 0, stream>>>(kvb, lc2_w, lc2_b, vnew, L, Lr);
    k_attn<<<BB * 2 * QT_TILES, 32, 0, stream>>>(qlin, kvb, vnew, xcat, Lm, 2, 64);
  }

  // out = x_q + (xcat @ proj_w^T + proj_b), scattered to (img, msg) tuple layout
  {
    dim3 g(CC / 64, (BB * NTOK) / 16);
    k_gemm_wmma<<<g, 32, 0, stream>>>(xcat, proj_w, proj_b, im2c, BB * NTOK, CC, CC);
    k_resid_out<<<2048, 256, 0, stream>>>(im2c, xq, out);
  }
}
